// MLP_Pointwise_43989055046138
// MI455X (gfx1250) — matrix operand order repaired and checked by reading, not compiled, not run on hardware
//
#include <hip/hip_runtime.h>
#include <hip/hip_bf16.h>

typedef _Float16 f16;
typedef __attribute__((ext_vector_type(16))) _Float16 v16h;
typedef __attribute__((ext_vector_type(8)))  _Float16 v8h;
typedef __attribute__((ext_vector_type(4)))  _Float16 v4h;
typedef __attribute__((ext_vector_type(8)))  float    v8f;
typedef __attribute__((ext_vector_type(4)))  float    v4f;

#define N_TOTAL  65536
#define D_INDIM  1024
#define HDIM     128
#define C_OUTCH  16
#define CG       4     // channels per workgroup
#define MT       64    // rows per workgroup
#define KC       32    // K chunk = one WMMA K window
#define NTHREADS 256

// Element (k-within-32 kk, column n) of a [32 x 128] B tile -> WMMA-B operand
// layout: col-block cb = n/16, lane slot = n%16 + 16*(kk>=16), 16 contiguous
// f16 per lane with K ascending.
__device__ __forceinline__ int bidx(int n, int kk) {
    return (((n >> 4) * 32 + ((n & 15) | (kk & 16))) << 4) | (kk & 15);
}

// Convert a 4k x 4n f32 block (row-major, stride srcStride) into the swizzled
// f16 B layout. Each column's 4 K-values are contiguous -> one ds_store_b64.
__device__ __forceinline__ void fill_b_tile(f16* __restrict__ bs,
                                            const float* __restrict__ src,
                                            int srcStride, int kb, int nb) {
    v4f r0 = *(const v4f*)(src + (size_t)(kb + 0) * srcStride + nb);
    v4f r1 = *(const v4f*)(src + (size_t)(kb + 1) * srcStride + nb);
    v4f r2 = *(const v4f*)(src + (size_t)(kb + 2) * srcStride + nb);
    v4f r3 = *(const v4f*)(src + (size_t)(kb + 3) * srcStride + nb);
    #pragma unroll
    for (int ni = 0; ni < 4; ++ni) {
        v4h p = { (f16)r0[ni], (f16)r1[ni], (f16)r2[ni], (f16)r3[ni] };
        *(v4h*)(bs + bidx(nb + ni, kb)) = p;
    }
}

__global__ __launch_bounds__(NTHREADS)
void mlp_pointwise_wmma(const float* __restrict__ x,
                        const float* __restrict__ W1,
                        const float* __restrict__ b1,
                        const float* __restrict__ W2,
                        const float* __restrict__ b2,
                        const float* __restrict__ W3,
                        const float* __restrict__ b3,
                        float* __restrict__ out)
{
    __shared__ __align__(16) f16 As[MT * KC];           //  4 KB: x chunk, f16 row-major
    __shared__ __align__(16) f16 Bs[CG * 8 * 32 * 16];  // 32 KB: weight chunks, swizzled
    __shared__ __align__(16) f16 Hs[MT * HDIM];         // 16 KB: h1/h2 of current channel
    __shared__ float red[4 * MT];                       //  1 KB: layer-3 reduction

    const int tid  = threadIdx.x;
    const int lane = tid & 31;
    const int wave = tid >> 5;
    const int g    = lane >> 4;   // lane half: selects A K-subgroups / B K-range
    const int ml   = lane & 15;   // M (A rows) or N (B/C cols) within a tile

    const int cgBase  = blockIdx.x * CG;   // fastest grid dim: channel group (L2 reuse of x)
    const int rowBase = blockIdx.y * MT;

    // Layer-1 wave tiling: wave owns ALL 4 row blocks x one col half of one channel.
    const int cc = wave & 3;    // channel within group
    const int hf = wave >> 2;   // column half (col blocks hf*4 .. hf*4+3)

    // ---------------- Layer 1: [64 x 1024] x 4x [1024 x 128] ----------------
    v8f acc[4][4] = {};   // [row block][col block within half]

    for (int kt = 0; kt < D_INDIM / KC; ++kt) {
        __syncthreads();  // previous chunk fully consumed

        // x chunk: 64 rows x 32 K (512 float4, 2/thread) -> packed v4h stores
        for (int i = tid; i < MT * KC / 4; i += NTHREADS) {
            int r = i >> 3, k4 = (i & 7) << 2;
            v4f v = *(const v4f*)(x + (size_t)(rowBase + r) * D_INDIM + kt * KC + k4);
            v4h p = { (f16)v[0], (f16)v[1], (f16)v[2], (f16)v[3] };
            *(v4h*)(As + r * KC + k4) = p;
        }
        // W1 chunks for 4 channels: 4x (32 x 128), 4k x 4n blocks, 4/thread
        for (int i = tid; i < CG * 256; i += NTHREADS) {
            int ch = i >> 8;
            int b  = i & 255;
            int kb = (b >> 5) << 2;
            int nb = (b & 31) << 2;
            const float* src = W1 + ((size_t)(cgBase + ch) * D_INDIM + kt * KC) * HDIM;
            fill_b_tile(Bs + ch * 4096, src, HDIM, kb, nb);
        }
        __syncthreads();

        // 4 B fragments held in regs, reused across 4 row blocks -> 16 WMMAs/chunk
        const f16* bsc = Bs + cc * 4096;
        v16h bfr[4];
        #pragma unroll
        for (int t = 0; t < 4; ++t)
            bfr[t] = *(const v16h*)(bsc + (((hf * 4 + t) * 32 + lane) << 4));
        #pragma unroll
        for (int rb = 0; rb < 4; ++rb) {
            union { v16h v; v8h h[2]; } a;
            const f16* ar = As + (rb * 16 + ml) * KC + g * 16;
            a.h[0] = *(const v8h*)(ar);
            a.h[1] = *(const v8h*)(ar + 8);
            #pragma unroll
            for (int t = 0; t < 4; ++t)
                acc[rb][t] = __builtin_amdgcn_wmma_f32_16x16x32_f16(
                    false, a.v, false, bfr[t], (short)0, acc[rb][t], false, false);
        }
    }

    // ---------------- Layers 2 & 3, one channel of the group at a time ----------------
    for (int cc2 = 0; cc2 < CG; ++cc2) {
        const int c = cgBase + cc2;
        __syncthreads();  // Hs / red free from previous channel

        // h1 = relu(acc + b1) -> Hs  (only the 2 waves owning this channel)
        if (cc == cc2) {
            const float* b1c = b1 + (size_t)c * HDIM;
            #pragma unroll
            for (int rb = 0; rb < 4; ++rb)
                #pragma unroll
                for (int t = 0; t < 4; ++t) {
                    int n = (hf * 4 + t) * 16 + ml;
                    float bias = b1c[n];
                    #pragma unroll
                    for (int v = 0; v < 8; ++v)
                        Hs[(rb * 16 + v + 8 * g) * HDIM + n] =
                            (f16)fmaxf(acc[rb][t][v] + bias, 0.f);
                }
        }

        // Layer 2: [64 x 128] x [128 x 128]; wave tiling: (row block, col half)
        const int rb2 = wave & 3, hf2 = wave >> 2;
        v8f acc2[4] = {};
        for (int kt = 0; kt < HDIM / KC; ++kt) {
            __syncthreads();  // (kt=0: also publishes h1) Bs free from prev use
            {   // W2 chunk: 32 x 128, 256 blocks -> exactly 1 per thread
                int kb = (tid >> 5) << 2;
                int nb = (tid & 31) << 2;
                const float* src = W2 + ((size_t)c * HDIM + kt * KC) * HDIM;
                fill_b_tile(Bs, src, HDIM, kb, nb);
            }
            __syncthreads();

            v16h bfr[4];
            #pragma unroll
            for (int t = 0; t < 4; ++t)
                bfr[t] = *(const v16h*)(Bs + (((hf2 * 4 + t) * 32 + lane) << 4));
            union { v16h v; v8h h[2]; } a;
            const f16* ar = Hs + (rb2 * 16 + ml) * HDIM + kt * KC + g * 16;
            a.h[0] = *(const v8h*)(ar);
            a.h[1] = *(const v8h*)(ar + 8);
            #pragma unroll
            for (int t = 0; t < 4; ++t)
                acc2[t] = __builtin_amdgcn_wmma_f32_16x16x32_f16(
                    false, a.v, false, bfr[t], (short)0, acc2[t], false, false);
        }
        __syncthreads();  // all waves done reading h1

        // h2 = relu(acc2 + b2) -> Hs
        {
            const float* b2c = b2 + (size_t)c * HDIM;
            #pragma unroll
            for (int t = 0; t < 4; ++t) {
                int n = (hf2 * 4 + t) * 16 + ml;
                float bias = b2c[n];
                #pragma unroll
                for (int v = 0; v < 8; ++v)
                    Hs[(rb2 * 16 + v + 8 * g) * HDIM + n] =
                        (f16)fmaxf(acc2[t][v] + bias, 0.f);
            }
        }
        __syncthreads();

        // Layer 3: out[n, c] = dot(h2[n, :], W3[c]) + b3[c]
        {
            int r = tid & 63, q = tid >> 6;
            const f16*  hrow = Hs + r * HDIM + q * 32;
            const float* W3c = W3 + (size_t)c * HDIM;
            float s = 0.f;
            #pragma unroll
            for (int k = 0; k < 32; ++k) s += (float)hrow[k] * W3c[q * 32 + k];
            red[q * MT + r] = s;
        }
        __syncthreads();
        if (tid < MT) {
            float s = red[tid] + red[MT + tid] + red[2 * MT + tid] + red[3 * MT + tid]
                    + b3[c];
            out[(size_t)(rowBase + tid) * C_OUTCH + c] = s;
        }
    }
}

extern "C" void kernel_launch(void* const* d_in, const int* in_sizes, int n_in,
                              void* d_out, int out_size, void* d_ws, size_t ws_size,
                              hipStream_t stream) {
    const float* x  = (const float*)d_in[0];
    const float* W1 = (const float*)d_in[1];
    const float* b1 = (const float*)d_in[2];
    const float* W2 = (const float*)d_in[3];
    const float* b2 = (const float*)d_in[4];
    const float* W3 = (const float*)d_in[5];
    const float* b3 = (const float*)d_in[6];
    float* out = (float*)d_out;

    dim3 grid(C_OUTCH / CG, N_TOTAL / MT);  // x = channel group (fastest) -> L2 reuse of x tiles
    mlp_pointwise_wmma<<<grid, dim3(NTHREADS), 0, stream>>>(x, W1, b1, W2, b2, W3, b3, out);
}
